// NetG_3289944948771
// MI455X (gfx1250) — matrix operand order repaired and checked by reading, not compiled, not run on hardware
//
#include <hip/hip_runtime.h>
#include <hip/hip_bf16.h>

// ---------------------------------------------------------------------------
// Seq2seq GRU (encoder + decoder) for MI455X / gfx1250, wave32 + WMMA bf16.
//
// B=512, T=256, H=256, D=3.  Grid: 32 persistent WGs x 256 threads (8 waves).
// Each WG owns a 16-row batch tile; h tile lives in LDS (bf16).
//
// Weight-stationary: each wave holds its 48 W_hh B-fragments (6 N-tiles x
// 8 k-chunks, 384 VGPRs) in registers for the whole 256-step recurrence
// (gfx1250 waves address up to 1024 VGPRs).  Per step the critical path is
// only: 2x ds_load_b128 (A fragment of h) + 6 chained
// v_wmma_f32_16x16x32_bf16 per k-chunk, then fused GRU gate math,
// fused D=3 input projection, and (decoder) fused output projection.
// ---------------------------------------------------------------------------

typedef __attribute__((ext_vector_type(16))) __bf16 v16bf;
typedef __attribute__((ext_vector_type(8)))  __bf16 v8bf;
typedef __attribute__((ext_vector_type(8)))  float  v8f;

#define B_   512
#define T_   256
#define H_   256
#define MT   16            // batch rows per workgroup (WMMA M)
#define NWG  (B_ / MT)     // 32 workgroups
#define WAVES 8
#define HPAD 264           // hbuf row stride in bf16 (528 B, 16B aligned)

// Packed W_hh fragment layout (per matrix):
//   frag index f = ((wave*6 + idx6)*8 + k), idx6 = gate*2 + ntile
//   fragment = 32 lanes x 16 bf16 (contiguous per lane)
//   B-matrix N  = gate*256 + 32*wave + 16*ntile + (lane&15)
//   B-matrix K  = 32*k + 16*(lane>>4) + e,  e = 0..15
#define FRAGS_PER_MAT (WAVES * 6 * 8)          // 384
#define ELEMS_PER_MAT (FRAGS_PER_MAT * 512)    // 196608 == 3H*H

// ---------------------------------------------------------------------------
// Prep: convert W_hh (fp32, [3H][H] row-major) -> packed bf16 fragments in ws
// ---------------------------------------------------------------------------
__global__ void pack_whh_kernel(const float* __restrict__ Wenc,
                                const float* __restrict__ Wdec,
                                __bf16* __restrict__ out) {
    const int perMat = ELEMS_PER_MAT / 16;      // 12288 threads per matrix
    int tid = blockIdx.x * blockDim.x + threadIdx.x;
    if (tid >= 2 * perMat) return;
    int mat  = tid / perMat;
    int r    = tid - mat * perMat;
    int frag = r >> 5;
    int lane = r & 31;
    int w    = frag / 48;
    int rem  = frag - w * 48;
    int idx6 = rem >> 3;
    int k    = rem & 7;
    int g    = idx6 >> 1;
    int t    = idx6 & 1;
    int nrow = g * 256 + 32 * w + 16 * t + (lane & 15);   // W_hh row (gate out)
    int kbase = 32 * k + 16 * (lane >> 4);                // W_hh col (contract)
    const float* W = mat ? Wdec : Wenc;
    __bf16* dst = out + (size_t)mat * ELEMS_PER_MAT + (size_t)frag * 512 + lane * 16;
#pragma unroll
    for (int e = 0; e < 16; ++e)
        dst[e] = (__bf16)W[nrow * H_ + kbase + e];
}

// ---------------------------------------------------------------------------
// Main persistent kernel
// ---------------------------------------------------------------------------
__device__ __forceinline__ float sigf(float x) {
    return 1.0f / (1.0f + __expf(-x));
}
__device__ __forceinline__ float tanh_fast(float x) {
    float e = __expf(2.0f * x);
    return (e - 1.0f) / (e + 1.0f);
}

__device__ __forceinline__ v16bf cat16(v8bf lo, v8bf hi) {
    return __builtin_shufflevector(lo, hi, 0, 1, 2, 3, 4, 5, 6, 7,
                                   8, 9, 10, 11, 12, 13, 14, 15);
}

__global__ __launch_bounds__(256, 1) void gru_seq2seq_kernel(
    const float* __restrict__ Xp, const float* __restrict__ Xf,
    const float* __restrict__ noise,
    const float* __restrict__ Wih_e, const float* __restrict__ bih_e,
    const float* __restrict__ bhh_e,
    const float* __restrict__ Wih_d, const float* __restrict__ bih_d,
    const float* __restrict__ bhh_d,
    const float* __restrict__ Wfc,   const float* __restrict__ bfc,
    const __bf16* __restrict__ packW,
    float* __restrict__ out)
{
    __shared__ __align__(16) __bf16 hbuf[MT][HPAD];   // h tile, bf16
    __shared__ float xs[MT][4];                       // x_t tile (D=3)
    __shared__ float oacc[MT][4];                     // output-proj accum

    const int tid  = threadIdx.x;
    const int wave = tid >> 5;
    const int lane = tid & 31;
    const int nloc = lane & 15;
    const int hi   = lane >> 4;       // 0: M 0..7 / K-lo half, 1: M 8..15
    const int wg   = blockIdx.x;

    // zero h in LDS
    for (int i = tid; i < MT * HPAD; i += 256)
        ((__bf16*)hbuf)[i] = (__bf16)0.0f;

    float hreg[2][8];                 // lane's slice of h (C/D layout)
#pragma unroll
    for (int t = 0; t < 2; ++t)
#pragma unroll
        for (int v = 0; v < 8; ++v) hreg[t][v] = 0.0f;

    // per-lane W_fc columns (for fused decoder projection)
    float wfc[3][2];
#pragma unroll
    for (int d = 0; d < 3; ++d)
#pragma unroll
        for (int t = 0; t < 2; ++t)
            wfc[d][t] = Wfc[d * H_ + 32 * wave + 16 * t + nloc];

    for (int phase = 0; phase < 2; ++phase) {
        const float*  Wih = phase ? Wih_d : Wih_e;
        const float*  bih = phase ? bih_d : bih_e;
        const float*  bhh = phase ? bhh_d : bhh_e;
        const float*  X   = phase ? Xf : Xp;
        const __bf16* WB  = packW + (size_t)phase * ELEMS_PER_MAT;

        // ---- load this wave's W_hh B-fragments into registers (stationary
        //      for all 256 timesteps): 6 N-tiles x 8 k-chunks x 8 VGPRs ----
        v16bf bw[6][8];
#pragma unroll
        for (int i6 = 0; i6 < 6; ++i6) {
#pragma unroll
            for (int k = 0; k < 8; ++k) {
                const v8bf* bp = (const v8bf*)(
                    WB + (size_t)(((wave * 6 + i6) * 8 + k) * 512)
                       + lane * 16);
                bw[i6][k] = cat16(bp[0], bp[1]);
            }
        }

        // per-lane gate constants: W_ih rows (D=3) and biases
        float wi[6][3], bi6[6], bh6[6];
#pragma unroll
        for (int i6 = 0; i6 < 6; ++i6) {
            int g = i6 >> 1, t = i6 & 1;
            int grow = g * 256 + 32 * wave + 16 * t + nloc;
            wi[i6][0] = Wih[grow * 3 + 0];
            wi[i6][1] = Wih[grow * 3 + 1];
            wi[i6][2] = Wih[grow * 3 + 2];
            bi6[i6]   = bih[grow];
            bh6[i6]   = bhh[grow];
        }

        for (int ts = 0; ts < T_; ++ts) {
            __syncthreads();          // prior-step h writes visible

            // stage x_t tile (decoder: right-shifted, zero at ts==0)
            if (tid < 48) {
                int m = tid / 3, d = tid - 3 * m;
                int b = wg * MT + m;
                float xv;
                if (phase == 0) xv = X[(b * T_ + ts) * 3 + d];
                else            xv = (ts == 0) ? 0.0f
                                               : X[(b * T_ + ts - 1) * 3 + d];
                xs[m][d]   = xv;
                oacc[m][d] = 0.0f;
            }

            v8f acc[6] = {};          // six 16x16 f32 accumulators

#pragma unroll
            for (int k = 0; k < 8; ++k) {
                // A fragment: lane's 16 contiguous K of h (K = 32k + 16*hi + e, matches packed B)
                const v8bf* ap =
                    (const v8bf*)&hbuf[nloc][k * 32 + hi * 16];
                v16bf a = cat16(ap[0], ap[1]);   // +8 bf16 elements
#pragma unroll
                for (int i6 = 0; i6 < 6; ++i6) {
                    acc[i6] = __builtin_amdgcn_wmma_f32_16x16x32_bf16(
                        false, a, false, bw[i6][k], (short)0,
                        acc[i6], false, false);
                }
            }
            __syncthreads();          // all h reads done; xs/oacc visible

            // GRU gate math (each lane owns (m, col) elements in C/D layout)
#pragma unroll
            for (int t = 0; t < 2; ++t) {
                const int ir = t, iz = 2 + t, in_ = 4 + t;
#pragma unroll
                for (int v = 0; v < 8; ++v) {
                    int m = v + hi * 8;
                    float x0 = xs[m][0], x1 = xs[m][1], x2 = xs[m][2];
                    float xr = fmaf(wi[ir][0], x0,
                               fmaf(wi[ir][1], x1,
                               fmaf(wi[ir][2], x2, bi6[ir])));
                    float xz = fmaf(wi[iz][0], x0,
                               fmaf(wi[iz][1], x1,
                               fmaf(wi[iz][2], x2, bi6[iz])));
                    float xn = fmaf(wi[in_][0], x0,
                               fmaf(wi[in_][1], x1,
                               fmaf(wi[in_][2], x2, bi6[in_])));
                    float r = sigf(xr + acc[ir][v] + bh6[ir]);
                    float z = sigf(xz + acc[iz][v] + bh6[iz]);
                    float n = tanh_fast(xn + r * (acc[in_][v] + bh6[in_]));
                    float hn = (1.0f - z) * n + z * hreg[t][v];
                    hreg[t][v] = hn;
                    hbuf[m][32 * wave + 16 * t + nloc] = (__bf16)hn;
                }
            }

            if (phase == 1) {
                // fused output projection: partials -> butterfly -> ds_add
#pragma unroll
                for (int v = 0; v < 8; ++v) {
                    int m = v + hi * 8;
                    float p0 = hreg[0][v] * wfc[0][0] + hreg[1][v] * wfc[0][1];
                    float p1 = hreg[0][v] * wfc[1][0] + hreg[1][v] * wfc[1][1];
                    float p2 = hreg[0][v] * wfc[2][0] + hreg[1][v] * wfc[2][1];
#pragma unroll
                    for (int msk = 1; msk < 16; msk <<= 1) {
                        p0 += __shfl_xor(p0, msk, 32);
                        p1 += __shfl_xor(p1, msk, 32);
                        p2 += __shfl_xor(p2, msk, 32);
                    }
                    if (nloc == 0) {
                        atomicAdd(&oacc[m][0], p0);
                        atomicAdd(&oacc[m][1], p1);
                        atomicAdd(&oacc[m][2], p2);
                    }
                }
                __syncthreads();
                if (tid < 48) {
                    int m = tid / 3, d = tid - 3 * m;
                    out[((wg * MT + m) * T_ + ts) * 3 + d] =
                        oacc[m][d] + bfc[d];
                }
            }
        }

        if (phase == 0) {
            // hidden = h_T + noise  (noise: [1,B,H])
#pragma unroll
            for (int t = 0; t < 2; ++t)
#pragma unroll
                for (int v = 0; v < 8; ++v) {
                    int m    = v + hi * 8;
                    int hcol = 32 * wave + 16 * t + nloc;
                    float hn = hreg[t][v] + noise[(wg * MT + m) * H_ + hcol];
                    hreg[t][v]    = hn;
                    hbuf[m][hcol] = (__bf16)hn;
                }
            // decoder-loop top __syncthreads() publishes these writes
        }
    }
}

// ---------------------------------------------------------------------------
extern "C" void kernel_launch(void* const* d_in, const int* in_sizes, int n_in,
                              void* d_out, int out_size, void* d_ws,
                              size_t ws_size, hipStream_t stream) {
    (void)in_sizes; (void)n_in; (void)out_size; (void)ws_size;

    const float* Xp     = (const float*)d_in[0];
    const float* Xf     = (const float*)d_in[1];
    const float* noise  = (const float*)d_in[2];
    const float* Wih_e  = (const float*)d_in[3];
    const float* Whh_e  = (const float*)d_in[4];
    const float* bih_e  = (const float*)d_in[5];
    const float* bhh_e  = (const float*)d_in[6];
    const float* Wih_d  = (const float*)d_in[7];
    const float* Whh_d  = (const float*)d_in[8];
    const float* bih_d  = (const float*)d_in[9];
    const float* bhh_d  = (const float*)d_in[10];
    const float* Wfc    = (const float*)d_in[11];
    const float* bfc    = (const float*)d_in[12];
    float*       out    = (float*)d_out;
    __bf16*      packW  = (__bf16*)d_ws;   // 2 * 384 KB

    const int prepThreads = 2 * (ELEMS_PER_MAT / 16);
    pack_whh_kernel<<<(prepThreads + 255) / 256, 256, 0, stream>>>(
        Whh_e, Whh_d, packW);

    gru_seq2seq_kernel<<<NWG, 256, 0, stream>>>(
        Xp, Xf, noise,
        Wih_e, bih_e, bhh_e,
        Wih_d, bih_d, bhh_d,
        Wfc, bfc, packW, out);
}
